// I2GNN_7696581394713
// MI455X (gfx1250) — matrix operand order repaired and checked by reading, not compiled, not run on hardware
//
#include <hip/hip_runtime.h>

typedef _Float16 h8   __attribute__((ext_vector_type(8)));
typedef _Float16 v16h __attribute__((ext_vector_type(16)));
typedef float    v8f  __attribute__((ext_vector_type(8)));

#define EMB 128
#define TUP_BLK 128      // tuples per block (= 16 pairs = 2 nodes)
#define PAIRS_BLK 16
#define S_WIN 8
#define MAX_NC 32

// ---------------- WMMA helpers (CDNA5 wave32, 16x16x32 f16 -> f32) ----------

__device__ __forceinline__ v16h make_v16(h8 lo, h8 hi) {
    v16h r;
#pragma unroll
    for (int i = 0; i < 8; ++i) { r[i] = lo[i]; r[i + 8] = hi[i]; }
    return r;
}

// A-matrix 16x32 f16 fragment. Lanes 0-15: M=lane, K in {kbase+0..7, kbase+16..23}
// with kbase = (lane>>4)*8 per ISA 7.12.2.
__device__ __forceinline__ v16h load_a_frag(const _Float16* A, int mRow, int ks, int lane) {
    const _Float16* p = A + mRow * EMB + ks * 32 + ((lane >> 4) << 3);
    h8 lo = *(const h8*)(p);
    h8 hi = *(const h8*)(p + 16);
    return make_v16(lo, hi);
}

// B-matrix 32x16 f16 fragment from transposed weights Wt[N][K].
// Lane n = L&15, K in {kbase+0..7, kbase+16..23} with kbase = (L>>4)*8.
__device__ __forceinline__ v16h load_b_frag(const _Float16* Wt, int ntile, int ks, int lane) {
    const _Float16* p = Wt + (ntile * 16 + (lane & 15)) * EMB + ks * 32 + ((lane >> 4) << 3);
    h8 lo = *(const h8*)(p);
    h8 hi = *(const h8*)(p + 16);
    return make_v16(lo, hi);
}

// One wave computes a 16x128 strip of Z = A[128x128] @ W[128x128] (W given transposed).
__device__ __forceinline__ void wave_gemm_128(const _Float16* A, const _Float16* Wt,
                                              int wm, int lane, v8f acc[8]) {
#pragma unroll
    for (int n = 0; n < 8; ++n)
#pragma unroll
        for (int e = 0; e < 8; ++e) acc[n][e] = 0.f;

    const int aRow = wm * 16 + (lane & 15);
#pragma unroll
    for (int ks = 0; ks < 4; ++ks) {
        v16h a = load_a_frag(A, aRow, ks, lane);
#pragma unroll
        for (int n = 0; n < 8; ++n) {
            v16h b = load_b_frag(Wt, n, ks, lane);
            acc[n] = __builtin_amdgcn_wmma_f32_16x16x32_f16(
                false, a, false, b, (short)0, acc[n], false, false);
        }
    }
}

// ---------------- Kernel 1: xe = emb_x[x] gather --------------------------------

__global__ __launch_bounds__(256) void gather_xe_kernel(const int* __restrict__ x,
                                                        const float* __restrict__ emb_x,
                                                        float* __restrict__ xe, int total) {
    int i = blockIdx.x * 256 + threadIdx.x;
    if (i < total) {
        int n = i >> 7, f = i & 127;
        xe[i] = emb_x[x[n] * EMB + f];
    }
}

// ---------------- Kernel 2: lin = xe @ w_ti + b_ti (WMMA) -----------------------

__global__ __launch_bounds__(256) void lin_gemm_kernel(const float* __restrict__ xe,
                                                       const float* __restrict__ w_ti,
                                                       const float* __restrict__ b_ti,
                                                       float* __restrict__ lin) {
    extern __shared__ char smem[];
    _Float16* Axe = (_Float16*)smem;            // 128x128 f16 = 32 KB
    _Float16* Wt  = (_Float16*)(smem + 32768);  // 128x128 f16 = 32 KB

    const int tid = threadIdx.x, lane = tid & 31, wave = tid >> 5;
    const int rowBase = blockIdx.x * 128;

    for (int i = tid; i < EMB * EMB; i += 256) {
        Axe[i] = (_Float16)xe[rowBase * EMB + i];
        int o = i >> 7, in = i & 127;
        Wt[o * EMB + in] = (_Float16)w_ti[in * EMB + o];
    }
    __syncthreads();

    v8f acc[8];
    wave_gemm_128(Axe, Wt, wave, lane, acc);

#pragma unroll
    for (int n = 0; n < 8; ++n) {
        int col = n * 16 + (lane & 15);
        float bv = b_ti[col];
#pragma unroll
        for (int r = 0; r < 8; ++r) {
            int m = wave * 16 + r + ((lane >> 4) << 3);
            lin[(rowBase + m) * EMB + col] = acc[n][r] + bv;
        }
    }
}

// ---------------- Kernel 3: fused init + 5 conv layers + pooling ----------------
// Block = 128 tuples (16 pairs, 2 root nodes). Xv lives only in LDS.

__global__ __launch_bounds__(256) void mega_kernel(
    const float* __restrict__ xe, const float* __restrict__ lin,
    const int* __restrict__ tupleid0, const int* __restrict__ tupleid1,
    const int* __restrict__ tuplefeat, const int* __restrict__ msg_src,
    const int* __restrict__ msg_dst, const int* __restrict__ msg_edge,
    const int* __restrict__ edge_attr, const float* __restrict__ emb_ea,
    const float* __restrict__ emb_tf, const float* __restrict__ w_conv,
    const float* __restrict__ b_conv, float* __restrict__ nodeOut,
    int nc, int layers) {
    extern __shared__ char smem[];
    float*    Xv    = (float*)smem;                    //  64 KB  [128][128] f32
    _Float16* Agg   = (_Float16*)(smem + 65536);       //  32 KB  [128][128] f16
    _Float16* Wt    = (_Float16*)(smem + 98304);       //  32 KB  [128][128] f16 (transposed)
    float*    EmbEa = (float*)(smem + 131072);         //   8 KB  [16][128]
    float*    EmbTf = (float*)(smem + 139264);         //   4 KB  [16][64]
    unsigned* Msg   = (unsigned*)(smem + 143360);      //   2 KB  packed messages

    const int tid = threadIdx.x, lane = tid & 31, wave = tid >> 5;
    const int blk = blockIdx.x;
    const int tupBase = blk * TUP_BLK;
    const int pairBase = blk * PAIRS_BLK;

    // ---- stage embedding tables + packed messages ----
    for (int i = tid; i < 16 * EMB; i += 256) EmbEa[i] = emb_ea[i];
    for (int i = tid; i < 16 * 64; i += 256) EmbTf[i] = emb_tf[i];
    const int nmsg = PAIRS_BLK * nc;
    for (int i = tid; i < nmsg; i += 256) {
        int m = pairBase * nc + i;
        int d = msg_dst[m] - tupBase;                  // local dst tuple (k = d&7)
        int s = msg_src[m] - tupBase;                  // local src tuple
        int attr = edge_attr[msg_edge[m]];             // edge-attr index into EmbEa
        Msg[i] = ((unsigned)(d & 127) << 16) | ((unsigned)(s & 127) << 8) |
                 (unsigned)(attr & 255);
    }
    __syncthreads();

    // ---- tupleinit: Xv = xe[t0] * lin[t1] * emb_tf[tuplefeat]  (in LDS) ----
    {
        int tl = tid >> 1, h = tid & 1;                // tuple-local, feature half
        int gt = tupBase + tl;
        int t0 = tupleid0[gt], t1 = tupleid1[gt];
        int tfi = tuplefeat[2 * gt + h];
        const float4* xr = (const float4*)(xe + t0 * EMB + h * 64);
        const float4* lr = (const float4*)(lin + t1 * EMB + h * 64);
        const float4* tr = (const float4*)(EmbTf + tfi * 64);
        float4* dst = (float4*)(Xv + tl * EMB + h * 64);
#pragma unroll
        for (int j = 0; j < 16; ++j) {
            float4 a = xr[j], b = lr[j], c = tr[j];
            dst[j] = make_float4(a.x * b.x * c.x, a.y * b.y * c.y,
                                 a.z * b.z * c.z, a.w * b.w * c.w);
        }
    }
    __syncthreads();

    // ---- 5 conv layers, entirely in LDS ----
    for (int l = 0; l < layers; ++l) {
        // stage this layer's W transposed as f16
        const float* w = w_conv + l * EMB * EMB;
        for (int i = tid; i < EMB * EMB; i += 256) {
            int o = i >> 7, in = i & 127;
            Wt[o * EMB + in] = (_Float16)w[in * EMB + o];
        }

        // message pass: each thread owns (tuple, 64-feature half); messages for a
        // pair are contiguous in Msg, dst stays inside the pair's 8 tuples.
        {
            int tl = tid >> 1, h = tid & 1;
            int pl = tl >> 3, myk = tl & 7;
            float4 acc4[16];
#pragma unroll
            for (int j = 0; j < 16; ++j) acc4[j] = make_float4(0.f, 0.f, 0.f, 0.f);
            for (int c = 0; c < nc; ++c) {
                unsigned u = Msg[pl * nc + c];
                if ((int)((u >> 16) & 7u) == myk) {
                    int s = (u >> 8) & 255u;
                    int attr = u & 255u;
                    const float4* xs = (const float4*)(Xv + s * EMB + h * 64);
                    const float4* er = (const float4*)(EmbEa + attr * EMB + h * 64);
#pragma unroll
                    for (int j = 0; j < 16; ++j) {
                        float4 a = xs[j], e = er[j];
                        acc4[j].x += a.x * e.x; acc4[j].y += a.y * e.y;
                        acc4[j].z += a.z * e.z; acc4[j].w += a.w * e.w;
                    }
                }
            }
            h8* ag = (h8*)(Agg + tl * EMB + h * 64);
#pragma unroll
            for (int j = 0; j < 8; ++j) {
                h8 v;
                float4 a = acc4[2 * j], b = acc4[2 * j + 1];
                v[0] = (_Float16)a.x; v[1] = (_Float16)a.y;
                v[2] = (_Float16)a.z; v[3] = (_Float16)a.w;
                v[4] = (_Float16)b.x; v[5] = (_Float16)b.y;
                v[6] = (_Float16)b.z; v[7] = (_Float16)b.w;
                ag[j] = v;
            }
        }
        __syncthreads();

        // GEMM: Z = Agg @ W, then Xv += relu(Z + b)
        v8f acc[8];
        wave_gemm_128(Agg, Wt, wave, lane, acc);
        const float* bias = b_conv + l * EMB;
#pragma unroll
        for (int n = 0; n < 8; ++n) {
            int col = n * 16 + (lane & 15);
            float bv = bias[col];
#pragma unroll
            for (int r = 0; r < 8; ++r) {
                int m = wave * 16 + r + ((lane >> 4) << 3);
                float z = acc[n][r] + bv;
                z = z > 0.f ? z : 0.f;
                Xv[m * EMB + col] += z;
            }
        }
        __syncthreads();
    }

    // ---- fused pooling: max over k (8 tuples/pair), sum over j (8 pairs/node) ----
    {
        int nl = tid >> 7, f = tid & 127;              // local node (0..1), feature
        float ssum = 0.f;
        for (int pj = 0; pj < 8; ++pj) {
            float mx = -3.402823466e38f;
#pragma unroll
            for (int k = 0; k < S_WIN; ++k) {
                float v = Xv[(nl * 64 + pj * 8 + k) * EMB + f];
                mx = v > mx ? v : mx;
            }
            ssum += mx;
        }
        nodeOut[(blk * 2 + nl) * EMB + f] = ssum;
    }
}

// ---------------- Kernel 4: per-graph reduction + prediction --------------------

__global__ __launch_bounds__(128) void predict_kernel(const float* __restrict__ node,
                                                      const float* __restrict__ w_pred,
                                                      const float* __restrict__ b_pred,
                                                      float* __restrict__ out,
                                                      int nodesPerGraph) {
    __shared__ float red[128];
    int g = blockIdx.x, f = threadIdx.x;
    float s = 0.f;
    for (int j = 0; j < nodesPerGraph; ++j)
        s += node[(g * nodesPerGraph + j) * EMB + f];
    red[f] = s * w_pred[f];
    __syncthreads();
#pragma unroll
    for (int off = 64; off > 0; off >>= 1) {
        if (f < off) red[f] += red[f + off];
        __syncthreads();
    }
    if (f == 0) out[g] = red[0] + b_pred[0];
}

// ---------------- Launch --------------------------------------------------------

extern "C" void kernel_launch(void* const* d_in, const int* in_sizes, int n_in,
                              void* d_out, int out_size, void* d_ws, size_t ws_size,
                              hipStream_t stream) {
    const int*   x         = (const int*)d_in[0];
    const int*   edge_attr = (const int*)d_in[1];
    const int*   tuplefeat = (const int*)d_in[2];
    const int*   tupleid0  = (const int*)d_in[4];
    const int*   tupleid1  = (const int*)d_in[5];
    const int*   msg_src   = (const int*)d_in[6];
    const int*   msg_dst   = (const int*)d_in[7];
    const int*   msg_edge  = (const int*)d_in[8];
    const float* emb_x     = (const float*)d_in[12];
    const float* emb_ea    = (const float*)d_in[13];
    const float* emb_tf    = (const float*)d_in[14];
    const float* w_ti      = (const float*)d_in[15];
    const float* b_ti      = (const float*)d_in[16];
    const float* w_conv    = (const float*)d_in[17];
    const float* b_conv    = (const float*)d_in[18];
    const float* w_pred    = (const float*)d_in[19];
    const float* b_pred    = (const float*)d_in[20];

    const int N = in_sizes[0];          // 4096 nodes
    const int T = in_sizes[4];          // 262144 tuples
    const int P = in_sizes[10];         // 32768 pairs
    const int M = in_sizes[6];          // messages
    const int nc = M / P;               // 26 messages per pair
    const int layers = in_sizes[17] / (EMB * EMB);   // 5
    const int G = out_size;             // 64 graphs (output is [G,1])
    const int nodesPerGraph = N / G;    // 64

    float* ws_xe   = (float*)d_ws;              // [N,128]
    float* ws_lin  = ws_xe + (size_t)N * EMB;   // [N,128]
    float* ws_node = ws_lin + (size_t)N * EMB;  // [N,128]
    float* out = (float*)d_out;

    // 1) xe = emb_x[x]
    gather_xe_kernel<<<(N * EMB + 255) / 256, 256, 0, stream>>>(x, emb_x, ws_xe, N * EMB);

    // 2) lin = xe @ w_ti + b_ti   (WMMA)
    lin_gemm_kernel<<<N / 128, 256, 65536, stream>>>(ws_xe, w_ti, b_ti, ws_lin);

    // 3) fused tupleinit + 5 layers + pooling (WMMA, Xv LDS-resident)
    const size_t megaSmem = 145408;  // 142 KB carved as documented above
    mega_kernel<<<T / TUP_BLK, 256, megaSmem, stream>>>(
        ws_xe, ws_lin, tupleid0, tupleid1, tuplefeat, msg_src, msg_dst, msg_edge,
        edge_attr, emb_ea, emb_tf, w_conv, b_conv, ws_node, nc, layers);

    // 4) graph pooling + prediction
    predict_kernel<<<G, 128, 0, stream>>>(ws_node, w_pred, b_pred, out, nodesPerGraph);
}
